// SDGraphEncoder_2413771621112
// MI455X (gfx1250) — compile-verified, matrix operand order checked by reading, not run on hardware
//
#include <hip/hip_runtime.h>

typedef __bf16 bf16;
typedef __attribute__((ext_vector_type(16))) __bf16 bf16x16;
typedef __attribute__((ext_vector_type(8)))  __bf16 bf16x8;
typedef __attribute__((ext_vector_type(8)))  float  f32x8;

#define RSQF 0.9999950000374997f   // 1/sqrt(1+1e-5)
#define LSLOPE 0.2f

union V16U { bf16x16 v; bf16x8 h[2]; };

__device__ inline bf16x16 frag_ld(const bf16* rowbase, int laneHalf) {
  // gfx1250 16-bit A/B fragment: laneHalf==0 -> K {0..7,16..23}; laneHalf==1 -> K {8..15,24..31}
  V16U u;
  u.h[0] = *(const bf16x8*)(rowbase + laneHalf * 8);
  u.h[1] = *(const bf16x8*)(rowbase + 16 + laneHalf * 8);
  return u.v;
}

// CDNA5 async global->LDS copy (16B per lane), tracked by ASYNCcnt (ISA 15.18.3 op 98)
__device__ inline void async_cp16(const void* g, void* l) {
  unsigned lds = (unsigned)(size_t)l;  // LDS address = low 32 bits of the pointer
  asm volatile("global_load_async_to_lds_b128 %0, %1, off"
               :: "v"(lds), "v"(g) : "memory");
}
__device__ inline void wait_async0() {
  asm volatile("s_wait_asynccnt 0x0" ::: "memory");
}

// ---------------------------------------------------------------------------
// Generic batched bf16 GEMM: C[M,N] = A[M,K] (row-major, lda) * B[K,N] (row-major, ldb)
// optional epilogue: f = f*scale[m]*RSQF + bias[m]; leaky(0.2). K must be multiple of 32.
// ---------------------------------------------------------------------------
template <bool EPI, bool OUTBF>
__global__ __launch_bounds__(128) void gemm_bf16_k(
    const bf16* __restrict__ A, int lda, long long sA,
    const bf16* __restrict__ B, int ldb, long long sB,
    void* __restrict__ Cv, int ldc, long long sC,
    int M, int N, int K,
    const float* __restrict__ scale, const float* __restrict__ bias) {
  __shared__ bf16 As[64][40];  // stride 40 (80B) to dodge bank conflicts
  __shared__ bf16 Bs[64][40];  // stored transposed: [n][k]
  const int bz = blockIdx.z;
  A += (size_t)bz * sA;
  B += (size_t)bz * sB;
  const int m0 = blockIdx.y * 64, n0 = blockIdx.x * 64;
  const int tid = threadIdx.x;
  const int lane = tid & 31, wave = tid >> 5;
  const int laneHalf = lane >> 4, lane16 = lane & 15;
  const int wm = (wave >> 1) * 32, wn = (wave & 1) * 32;

  f32x8 c00 = {}, c01 = {}, c10 = {}, c11 = {};

  for (int k0 = 0; k0 < K; k0 += 32) {
    __syncthreads();
    {  // stage A: 64 rows x 32 k, vectorized 32B per thread-half
      int r = tid >> 1, ks = (tid & 1) * 16;
      uint4 v0 = {0, 0, 0, 0}, v1 = {0, 0, 0, 0};
      if (m0 + r < M) {
        const bf16* src = A + (size_t)(m0 + r) * lda + k0 + ks;
        v0 = *(const uint4*)src;
        v1 = *(const uint4*)(src + 8);
      }
      *(uint4*)&As[r][ks] = v0;
      *(uint4*)&As[r][ks + 8] = v1;
    }
    {  // stage B: coalesced read along n, transposed store to [n][k]
      int kk = tid >> 2, nseg = (tid & 3) * 16;
      bf16 tmp[16];
      const bf16* src = B + (size_t)(k0 + kk) * ldb + n0 + nseg;
      if (n0 + nseg + 15 < N) {
        *(uint4*)tmp = *(const uint4*)src;
        *(uint4*)(tmp + 8) = *(const uint4*)(src + 8);
      } else {
#pragma unroll
        for (int j = 0; j < 16; ++j)
          tmp[j] = (n0 + nseg + j < N) ? src[j] : (bf16)0.0f;
      }
#pragma unroll
      for (int j = 0; j < 16; ++j) Bs[nseg + j][kk] = tmp[j];
    }
    __syncthreads();
    bf16x16 a0 = frag_ld(&As[wm + lane16][0], laneHalf);
    bf16x16 a1 = frag_ld(&As[wm + 16 + lane16][0], laneHalf);
    bf16x16 b0 = frag_ld(&Bs[wn + lane16][0], laneHalf);
    bf16x16 b1 = frag_ld(&Bs[wn + 16 + lane16][0], laneHalf);
    c00 = __builtin_amdgcn_wmma_f32_16x16x32_bf16(false, a0, false, b0, (short)0, c00, false, false);
    c01 = __builtin_amdgcn_wmma_f32_16x16x32_bf16(false, a0, false, b1, (short)0, c01, false, false);
    c10 = __builtin_amdgcn_wmma_f32_16x16x32_bf16(false, a1, false, b0, (short)0, c10, false, false);
    c11 = __builtin_amdgcn_wmma_f32_16x16x32_bf16(false, a1, false, b1, (short)0, c11, false, false);
  }

  char* Cb = (char*)Cv + (size_t)bz * sC * (OUTBF ? 2 : 4);
#pragma unroll
  for (int t = 0; t < 4; ++t) {
    f32x8 cc = (t == 0) ? c00 : (t == 1) ? c01 : (t == 2) ? c10 : c11;
    int ti = t >> 1, tj = t & 1;
#pragma unroll
    for (int r = 0; r < 8; ++r) {
      int gm = m0 + wm + ti * 16 + 8 * laneHalf + r;  // D: m = r + 8*(lane>=16)
      int gn = n0 + wn + tj * 16 + lane16;            //    n = lane & 15
      if (gm < M && gn < N) {
        float f = cc[r];
        if constexpr (EPI) {
          f = f * (scale[gm] * RSQF) + bias[gm];
          f = f > 0.f ? f : LSLOPE * f;
        }
        if constexpr (OUTBF)
          ((bf16*)Cb)[(size_t)gm * ldc + gn] = (bf16)f;
        else
          ((float*)Cb)[(size_t)gm * ldc + gn] = f;
      }
    }
  }
}

// ---------------------------------------------------------------------------
// NT GEMM with async double-buffered staging: C[M,N] = A[M,K] * Bt[N,K]^T
// Both operands row-major with K contiguous -> staged via global_load_async_to_lds.
// Block tile 64(M) x 128(N), wave tile 32x64 (8 WMMAs/k-step). f32 out, no epilogue.
// Used for the dominant GEMMs: gram (X·X^T) and Y1/Y2 (Wd·X with Bt = x[n][c]).
// ---------------------------------------------------------------------------
__global__ __launch_bounds__(128) void gemm_bf16_nt_async_k(
    const bf16* __restrict__ A, int lda, long long sA,
    const bf16* __restrict__ Bt, int ldb, long long sB,
    float* __restrict__ C, int ldc, long long sC,
    int M, int N, int K) {
  __shared__ bf16 As[2][64][40];
  __shared__ bf16 Bs[2][128][40];
  const int bz = blockIdx.z;
  A  += (size_t)bz * sA;
  Bt += (size_t)bz * sB;
  C  += (size_t)bz * sC;
  const int m0 = blockIdx.y * 64, n0 = blockIdx.x * 128;
  const int tid = threadIdx.x, lane = tid & 31, wave = tid >> 5;
  const int laneHalf = lane >> 4, lane16 = lane & 15;
  const int wm = (wave >> 1) * 32, wn = (wave & 1) * 64;

  // issue async copies for one 64x32 (A) + 128x32 (B) k-slab into buffer `buf`
  auto stage = [&](int buf, int k0) {
#pragma unroll
    for (int c = 0; c < 6; ++c) {
      int chunk = c * 128 + tid;  // 256 A-chunks then 512 B-chunks, 16B each
      if (chunk < 256) {
        int row = chunk >> 2, seg = (chunk & 3) * 8;
        int gr = m0 + row; gr = gr < M ? gr : M - 1;   // clamp; OOB rows masked at store
        async_cp16(A + (size_t)gr * lda + k0 + seg, &As[buf][row][seg]);
      } else {
        int bc = chunk - 256;
        int row = bc >> 2, seg = (bc & 3) * 8;
        int gr = n0 + row; gr = gr < N ? gr : N - 1;
        async_cp16(Bt + (size_t)gr * ldb + k0 + seg, &Bs[buf][row][seg]);
      }
    }
  };

  f32x8 acc[2][4] = {};
  stage(0, 0);
  int cur = 0;
  for (int k0 = 0; k0 < K; k0 += 32) {
    wait_async0();       // this wave's async copies into `cur` complete
    __syncthreads();     // all waves' copies visible
    if (k0 + 32 < K) stage(cur ^ 1, k0 + 32);  // fill next buffer while computing
    if (k0 + 64 < K) {   // deepen pipeline: prefetch k+64 lines into cache
      int r = tid & 63;
      int gm = m0 + r; gm = gm < M ? gm : M - 1;
      int gn = n0 + tid; gn = gn < N ? gn : N - 1;
      __builtin_prefetch(A + (size_t)gm * lda + k0 + 64, 0, 0);
      __builtin_prefetch(Bt + (size_t)gn * ldb + k0 + 64, 0, 0);
    }
    bf16x16 a0 = frag_ld(&As[cur][wm + lane16][0], laneHalf);
    bf16x16 a1 = frag_ld(&As[cur][wm + 16 + lane16][0], laneHalf);
    bf16x16 b0 = frag_ld(&Bs[cur][wn + lane16][0], laneHalf);
    bf16x16 b1 = frag_ld(&Bs[cur][wn + 16 + lane16][0], laneHalf);
    bf16x16 b2 = frag_ld(&Bs[cur][wn + 32 + lane16][0], laneHalf);
    bf16x16 b3 = frag_ld(&Bs[cur][wn + 48 + lane16][0], laneHalf);
    acc[0][0] = __builtin_amdgcn_wmma_f32_16x16x32_bf16(false, a0, false, b0, (short)0, acc[0][0], false, false);
    acc[0][1] = __builtin_amdgcn_wmma_f32_16x16x32_bf16(false, a0, false, b1, (short)0, acc[0][1], false, false);
    acc[0][2] = __builtin_amdgcn_wmma_f32_16x16x32_bf16(false, a0, false, b2, (short)0, acc[0][2], false, false);
    acc[0][3] = __builtin_amdgcn_wmma_f32_16x16x32_bf16(false, a0, false, b3, (short)0, acc[0][3], false, false);
    acc[1][0] = __builtin_amdgcn_wmma_f32_16x16x32_bf16(false, a1, false, b0, (short)0, acc[1][0], false, false);
    acc[1][1] = __builtin_amdgcn_wmma_f32_16x16x32_bf16(false, a1, false, b1, (short)0, acc[1][1], false, false);
    acc[1][2] = __builtin_amdgcn_wmma_f32_16x16x32_bf16(false, a1, false, b2, (short)0, acc[1][2], false, false);
    acc[1][3] = __builtin_amdgcn_wmma_f32_16x16x32_bf16(false, a1, false, b3, (short)0, acc[1][3], false, false);
    cur ^= 1;
  }

#pragma unroll
  for (int i = 0; i < 2; ++i)
#pragma unroll
    for (int j = 0; j < 4; ++j)
#pragma unroll
      for (int r = 0; r < 8; ++r) {
        int gm = m0 + wm + i * 16 + 8 * laneHalf + r;
        int gn = n0 + wn + j * 16 + lane16;
        if (gm < M && gn < N) C[(size_t)gm * ldc + gn] = acc[i][j][r];
      }
}

// ---------------------------------------------------------------------------
// small helpers
// ---------------------------------------------------------------------------
__global__ void f32_to_bf16_k(const float* __restrict__ s, bf16* __restrict__ d, int n) {
  int i = blockIdx.x * 256 + threadIdx.x;
  if (i < n) d[i] = (bf16)s[i];
}

// spb[i][n] = sp[b,i,s], n = b*32+s  -> (128 x 256) bf16
__global__ void pack_sp_k(const float* __restrict__ sp, bf16* __restrict__ spb) {
  int i = blockIdx.x * 256 + threadIdx.x;
  if (i < 128 * 256) {
    int ch = i >> 8, n = i & 255, b = n >> 5, s = n & 31;
    spb[(size_t)ch * 256 + n] = (bf16)sp[((size_t)b * 128 + ch) * 32 + s];
  }
}

// attention: coef = softmax_o(leaky(bn((Wg@q)[o,n] - (Wg@kk)[b,o,s,p]))),
// us0[b,o,s] = v[o,n] * sum_p coef  (coef never materialized)
__global__ __launch_bounds__(256) void attn_kernel(
    const float* __restrict__ Aq, const float* __restrict__ vf,
    const float* __restrict__ G2, const float* __restrict__ gg,
    const float* __restrict__ bg, bf16* __restrict__ us0b) {
  int n = blockIdx.x;  // b*32+s
  int b = n >> 5, s = n & 31;
  int o = threadIdx.x;
  float aq = Aq[o * 256 + n];
  float vv = vf[o * 256 + n];
  float sc = gg[o] * RSQF, bo = bg[o];
  const float* g2 = G2 + ((size_t)b * 256 + o) * 4096 + s * 128;
  __shared__ float red[256];
  float acc = 0.f;
  for (int p = 0; p < 128; ++p) {
    float z = (aq - g2[p]) * sc + bo;
    z = z > 0.f ? z : LSLOPE * z;
    red[o] = z;
    __syncthreads();
#pragma unroll
    for (int st = 128; st > 0; st >>= 1) {
      if (o < st) red[o] = fmaxf(red[o], red[o + st]);
      __syncthreads();
    }
    float m = red[0];
    __syncthreads();
    float e = __expf(z - m);
    red[o] = e;
    __syncthreads();
#pragma unroll
    for (int st = 128; st > 0; st >>= 1) {
      if (o < st) red[o] += red[o + st];
      __syncthreads();
    }
    float ssum = red[0];
    __syncthreads();
    acc += e / ssum;
  }
  us0b[((size_t)b * 256 + o) * 32 + s] = (bf16)(vv * acc);
}

__device__ inline void top10_insert(float (&bv)[10], int (&bi)[10], float d, int m) {
  if (d > bv[9]) {  // strict: ties keep earlier index (matches jax top_k)
    bv[9] = d; bi[9] = m;
#pragma unroll
    for (int k = 9; k > 0; --k) {
      if (bv[k] > bv[k - 1]) {
        float tf = bv[k]; bv[k] = bv[k - 1]; bv[k - 1] = tf;
        int ti = bi[k]; bi[k] = bi[k - 1]; bi[k - 1] = ti;
      }
    }
  }
}

// sparse dgcnn knn: 32 points x 256 ch, all-pairs in LDS
__global__ __launch_bounds__(32) void sparse_topk_k(const bf16* __restrict__ x, int* __restrict__ idxo) {
  int b = blockIdx.x, s = threadIdx.x;
  __shared__ bf16 xs[256 * 32];
  const bf16* xb = x + (size_t)b * 256 * 32;
  for (int e = s; e < 256 * 32; e += 32) xs[e] = xb[e];
  __syncthreads();
  float xx_s = 0.f;
  for (int c = 0; c < 256; ++c) { float t = (float)xs[c * 32 + s]; xx_s += t * t; }
  __shared__ float xxs[32];
  xxs[s] = xx_s;
  __syncthreads();
  float bv[10]; int bi[10];
#pragma unroll
  for (int k = 0; k < 10; ++k) { bv[k] = -3.4e38f; bi[k] = 0; }
  for (int m = 0; m < 32; ++m) {
    float dot = 0.f;
    for (int c = 0; c < 256; ++c) dot += (float)xs[c * 32 + s] * (float)xs[c * 32 + m];
    top10_insert(bv, bi, 2.f * dot - xx_s - xxs[m], m);
  }
#pragma unroll
  for (int k = 0; k < 10; ++k) idxo[((size_t)b * 32 + s) * 10 + k] = bi[k];
}

// out1[b,o,s] = max_k leaky(bn(Z1[idx] - Z1[s] + Z2[s]))
__global__ void sparse_combine_k(const float* __restrict__ Z1, const float* __restrict__ Z2,
                                 const int* __restrict__ idx, const float* __restrict__ gs,
                                 const float* __restrict__ bs_, float* __restrict__ out) {
  int t = blockIdx.x * 256 + threadIdx.x;
  if (t >= 8 * 256 * 32) return;
  int s = t & 31, o = (t >> 5) & 255, b = t >> 13;
  size_t base = ((size_t)b * 256 + o) * 32;
  float z1n = Z1[base + s];
  float c2 = Z2[base + s] - z1n;
  float sc = gs[o] * RSQF, bb = bs_[o];
  const int* id = idx + ((size_t)b * 32 + s) * 10;
  float mx = -3.4e38f;
#pragma unroll
  for (int k = 0; k < 10; ++k) {
    float h = (Z1[base + id[k]] + c2) * sc + bb;
    h = h > 0.f ? h : LSLOPE * h;
    mx = fmaxf(mx, h);
  }
  out[t] = mx;
}

// union_dense pack: x[b,n,c] (nc) and x[b,c,n] (cn) bf16 + |x|^2
__global__ __launch_bounds__(64) void pack_dense_k(
    const float* __restrict__ dn, const float* __restrict__ sp,
    bf16* __restrict__ xnc, bf16* __restrict__ xcn, float* __restrict__ xx) {
  int n = blockIdx.x, b = blockIdx.y, t = threadIdx.x;
  int s = n >> 7;
  float loc = 0.f;
  for (int c = t; c < 256; c += 64) {
    float v = (c < 128) ? dn[((size_t)b * 128 + c) * 4096 + n]
                        : sp[((size_t)b * 128 + (c - 128)) * 32 + s];
    bf16 h = (bf16)v;
    xnc[((size_t)b * 4096 + n) * 256 + c] = h;
    xcn[((size_t)b * 256 + c) * 4096 + n] = h;
    float v2 = (float)h;
    loc += v2 * v2;
  }
  __shared__ float r[64];
  r[t] = loc;
  __syncthreads();
  for (int st = 32; st > 0; st >>= 1) { if (t < st) r[t] += r[t + st]; __syncthreads(); }
  if (t == 0) xx[(size_t)b * 4096 + n] = r[0];
}

// per-batch: d[n,m] = 2*gram[m,n] - xx[n] - xx[m]  (symmetry -> coalesced), top-10
__global__ __launch_bounds__(128) void dense_topk_k(
    const float* __restrict__ gram, const float* __restrict__ xx, int* __restrict__ idxo) {
  int n = blockIdx.x * 128 + threadIdx.x;
  float xn = xx[n];
  float bv[10]; int bi[10];
#pragma unroll
  for (int k = 0; k < 10; ++k) { bv[k] = -3.4e38f; bi[k] = 0; }
  for (int m = 0; m < 4096; ++m) {
    float d = 2.f * gram[(size_t)m * 4096 + n] - xn - xx[m];
    top10_insert(bv, bi, d, m);
  }
#pragma unroll
  for (int k = 0; k < 10; ++k) idxo[(size_t)n * 10 + k] = bi[k];
}

// ud[b,c,n] = max_k leaky(bn(Y1[idx] - Y1[n] + Y2[n]))
__global__ void dense_combine_k(const float* __restrict__ Y1, const float* __restrict__ Y2,
                                const int* __restrict__ idx, const float* __restrict__ gd,
                                const float* __restrict__ bd, float* __restrict__ ud) {
  int t = blockIdx.x * 256 + threadIdx.x;
  if (t >= 8 * 181 * 4096) return;
  int n = t & 4095;
  int rest = t >> 12;
  int c = rest % 181, b = rest / 181;
  size_t base = ((size_t)b * 181 + c) * 4096;
  float y1n = Y1[base + n];
  float c2 = Y2[base + n] - y1n;
  float sc = gd[c] * RSQF, bb = bd[c];
  const int* id = idx + ((size_t)b * 4096 + n) * 10;
  float mx = -3.4e38f;
#pragma unroll
  for (int k = 0; k < 10; ++k) {
    float h = (Y1[base + id[k]] + c2) * sc + bb;
    h = h > 0.f ? h : LSLOPE * h;
    mx = fmaxf(mx, h);
  }
  ud[t] = mx;
}

// 1x3 conv, stride (1,2), pad (1,1): out[b,o,s,w] over 181 in-ch, + bn + leaky
__global__ __launch_bounds__(1024) void conv_smp_k(
    const float* __restrict__ ud, const float* __restrict__ w,
    const float* __restrict__ g, const float* __restrict__ bb, float* __restrict__ out) {
  int bs = blockIdx.x;           // b*32 + s
  int b = bs >> 5, s = bs & 31;
  int o0 = blockIdx.y * 16;
  int wi = threadIdx.x;          // output w 0..63
  int oi = threadIdx.y;          // 0..15
  int t = oi * 64 + wi;
  __shared__ float uds[32][130];
  __shared__ float wsm[16][32][3];
  float acc = 0.f;
  for (int c0 = 0; c0 < 181; c0 += 32) {
    for (int e = t; e < 32 * 130; e += 1024) {
      int cc = e / 130, pp = e % 130, gp = pp - 1;
      float v = 0.f;
      if (c0 + cc < 181 && gp >= 0 && gp < 128)
        v = ud[((size_t)b * 181 + c0 + cc) * 4096 + s * 128 + gp];
      uds[cc][pp] = v;
    }
    for (int e = t; e < 16 * 32 * 3; e += 1024) {
      int oo = e / 96, cc = (e % 96) / 3, tt = e % 3;
      float v = 0.f;
      if (c0 + cc < 181) v = w[(((size_t)(o0 + oo) * 181) + c0 + cc) * 3 + tt];
      wsm[oo][cc][tt] = v;
    }
    __syncthreads();
#pragma unroll 4
    for (int cc = 0; cc < 32; ++cc) {
      float x0 = uds[cc][2 * wi];
      float x1 = uds[cc][2 * wi + 1];
      float x2 = uds[cc][2 * wi + 2];
      acc = fmaf(wsm[oi][cc][0], x0, acc);
      acc = fmaf(wsm[oi][cc][1], x1, acc);
      acc = fmaf(wsm[oi][cc][2], x2, acc);
    }
    __syncthreads();
  }
  int o = o0 + oi;
  float f = acc * (g[o] * RSQF) + bb[o];
  f = f > 0.f ? f : LSLOPE * f;
  out[((size_t)b * 256 + o) * 2048 + s * 64 + wi] = f;
}

// ---------------------------------------------------------------------------
extern "C" void kernel_launch(void* const* d_in, const int* in_sizes, int n_in,
                              void* d_out, int out_size, void* d_ws, size_t ws_size,
                              hipStream_t stream) {
  (void)in_sizes; (void)n_in; (void)out_size; (void)ws_size;
  const float* sp   = (const float*)d_in[0];
  const float* dn   = (const float*)d_in[1];
  const float* wq   = (const float*)d_in[2];
  const float* gq   = (const float*)d_in[3];
  const float* bq   = (const float*)d_in[4];
  const float* wk   = (const float*)d_in[5];
  const float* gk   = (const float*)d_in[6];
  const float* bk   = (const float*)d_in[7];
  const float* wv   = (const float*)d_in[8];
  const float* gv   = (const float*)d_in[9];
  const float* bv   = (const float*)d_in[10];
  const float* wg   = (const float*)d_in[11];
  const float* gg   = (const float*)d_in[12];
  const float* bg   = (const float*)d_in[13];
  const float* wsp  = (const float*)d_in[14];
  const float* gs   = (const float*)d_in[15];
  const float* bs_  = (const float*)d_in[16];
  const float* wd   = (const float*)d_in[17];
  const float* gd   = (const float*)d_in[18];
  const float* bd   = (const float*)d_in[19];
  const float* wsmp = (const float*)d_in[20];
  const float* gsmp = (const float*)d_in[21];
  const float* bsmp = (const float*)d_in[22];

  char* base = (char*)d_ws;
  size_t off = 0;
  auto alloc = [&](size_t bytes) -> void* {
    void* p = base + off;
    off = (off + bytes + 255) & ~(size_t)255;
    return p;
  };
  const long long BS = 8;
  bf16*  wq_b = (bf16*)alloc(256 * 128 * 2);
  bf16*  wk_b = (bf16*)alloc(256 * 128 * 2);
  bf16*  wv_b = (bf16*)alloc(256 * 128 * 2);
  bf16*  wg_b = (bf16*)alloc(256 * 256 * 2);
  bf16*  ws_b = (bf16*)alloc(256 * 512 * 2);
  bf16*  wd_b = (bf16*)alloc(181 * 512 * 2);
  bf16*  spb  = (bf16*)alloc(128 * 256 * 2);
  bf16*  dnb  = (bf16*)alloc((size_t)BS * 128 * 4096 * 2);
  bf16*  q_b  = (bf16*)alloc(256 * 256 * 2);
  float* v_f  = (float*)alloc(256 * 256 * 4);
  float* Aq   = (float*)alloc(256 * 256 * 4);
  bf16*  kkb  = (bf16*)alloc((size_t)BS * 256 * 4096 * 2);
  float* G2   = (float*)alloc((size_t)BS * 256 * 4096 * 4);
  bf16*  us0b = (bf16*)alloc((size_t)BS * 256 * 32 * 2);
  int*   idxs = (int*)alloc((size_t)BS * 32 * 10 * 4);
  float* Z1   = (float*)alloc((size_t)BS * 256 * 32 * 4);
  float* Z2   = (float*)alloc((size_t)BS * 256 * 32 * 4);
  bf16*  xnc  = (bf16*)alloc((size_t)BS * 4096 * 256 * 2);
  bf16*  xcn  = (bf16*)alloc((size_t)BS * 256 * 4096 * 2);
  float* xxd  = (float*)alloc((size_t)BS * 4096 * 4);
  int*   idxd = (int*)alloc((size_t)BS * 4096 * 10 * 4);
  float* Y1   = (float*)alloc((size_t)BS * 181 * 4096 * 4);
  float* Y2   = (float*)alloc((size_t)BS * 181 * 4096 * 4);
  float* ud   = (float*)alloc((size_t)BS * 181 * 4096 * 4);
  float* gram = (float*)alloc((size_t)4096 * 4096 * 4);

  auto cv = [&](const float* s, bf16* d, int n) {
    f32_to_bf16_k<<<(n + 255) / 256, 256, 0, stream>>>(s, d, n);
  };
  cv(wq, wq_b, 256 * 128); cv(wk, wk_b, 256 * 128); cv(wv, wv_b, 256 * 128);
  cv(wg, wg_b, 256 * 256); cv(wsp, ws_b, 256 * 512); cv(wd, wd_b, 181 * 512);
  cv(dn, dnb, 8 * 128 * 4096);
  pack_sp_k<<<(128 * 256 + 255) / 256, 256, 0, stream>>>(sp, spb);

  // q_act (bf16), v_act (f32): M=256, N=256(b*32+s), K=128
  gemm_bf16_k<true, true><<<dim3(4, 4, 1), 128, 0, stream>>>(
      wq_b, 128, 0, spb, 256, 0, q_b, 256, 0, 256, 256, 128, gq, bq);
  gemm_bf16_k<true, false><<<dim3(4, 4, 1), 128, 0, stream>>>(
      wv_b, 128, 0, spb, 256, 0, v_f, 256, 0, 256, 256, 128, gv, bv);
  // Aq = Wg @ q_act
  gemm_bf16_k<false, false><<<dim3(4, 4, 1), 128, 0, stream>>>(
      wg_b, 256, 0, q_b, 256, 0, Aq, 256, 0, 256, 256, 256, nullptr, nullptr);
  // kk_act (bf16, batched): M=256, N=4096, K=128
  gemm_bf16_k<true, true><<<dim3(64, 4, 8), 128, 0, stream>>>(
      wk_b, 128, 0, dnb, 4096, (long long)128 * 4096, kkb, 4096, (long long)256 * 4096,
      256, 4096, 128, gk, bk);
  // G2 = Wg @ kk_act
  gemm_bf16_k<false, false><<<dim3(64, 4, 8), 128, 0, stream>>>(
      wg_b, 256, 0, kkb, 4096, (long long)256 * 4096, G2, 4096, (long long)256 * 4096,
      256, 4096, 256, nullptr, nullptr);
  attn_kernel<<<256, 256, 0, stream>>>(Aq, v_f, G2, gg, bg, us0b);

  // sparse dgcnn
  sparse_topk_k<<<8, 32, 0, stream>>>(us0b, idxs);
  gemm_bf16_k<false, false><<<dim3(1, 4, 8), 128, 0, stream>>>(
      ws_b, 512, 0, us0b, 32, (long long)256 * 32, Z1, 32, (long long)256 * 32,
      256, 32, 256, nullptr, nullptr);
  gemm_bf16_k<false, false><<<dim3(1, 4, 8), 128, 0, stream>>>(
      ws_b + 256, 512, 0, us0b, 32, (long long)256 * 32, Z2, 32, (long long)256 * 32,
      256, 32, 256, nullptr, nullptr);
  sparse_combine_k<<<(8 * 256 * 32 + 255) / 256, 256, 0, stream>>>(
      Z1, Z2, idxs, gs, bs_, (float*)d_out);

  // dense dgcnn
  pack_dense_k<<<dim3(4096, 8), 64, 0, stream>>>(dn, sp, xnc, xcn, xxd);
  for (int b = 0; b < 8; ++b) {  // gram (X·X^T, NT async WMMA) + knn per batch
    gemm_bf16_nt_async_k<<<dim3(32, 64, 1), 128, 0, stream>>>(
        xnc + (size_t)b * 4096 * 256, 256, 0, xnc + (size_t)b * 4096 * 256, 256, 0,
        gram, 4096, 0, 4096, 4096, 256);
    dense_topk_k<<<32, 128, 0, stream>>>(gram, xxd + (size_t)b * 4096,
                                         idxd + (size_t)b * 4096 * 10);
  }
  // Y1 = Wd[:, :256] @ X, Y2 = Wd[:, 256:] @ X  (factored edge conv, Bt = xnc)
  gemm_bf16_nt_async_k<<<dim3(32, 3, 8), 128, 0, stream>>>(
      wd_b, 512, 0, xnc, 256, (long long)4096 * 256, Y1, 4096, (long long)181 * 4096,
      181, 4096, 256);
  gemm_bf16_nt_async_k<<<dim3(32, 3, 8), 128, 0, stream>>>(
      wd_b + 256, 512, 0, xnc, 256, (long long)4096 * 256, Y2, 4096, (long long)181 * 4096,
      181, 4096, 256);
  dense_combine_k<<<(8 * 181 * 4096 + 255) / 256, 256, 0, stream>>>(
      Y1, Y2, idxd, gd, bd, ud);
  conv_smp_k<<<dim3(256, 16), dim3(64, 16), 0, stream>>>(
      ud, wsmp, gsmp, bsmp, (float*)d_out + 65536);
}
